// Self_Attention_67173288509776
// MI455X (gfx1250) — matrix operand order repaired and checked by reading, not compiled, not run on hardware
//
#include <hip/hip_runtime.h>
#include <hip/hip_bf16.h>

// Problem constants (B, L, C, H) from the reference.
#define BB 8
#define LL 2048
#define CC 1024
#define HH 1024

typedef __attribute__((ext_vector_type(16))) __bf16 bf16x16;
typedef __attribute__((ext_vector_type(8)))  __bf16 bf16x8;
typedef __attribute__((ext_vector_type(8)))  float  f32x8;

union ABReg {
    bf16x8  h[2];
    bf16x16 v;
};

static __device__ __forceinline__ f32x8 fzero8() {
    f32x8 z = {0.f, 0.f, 0.f, 0.f, 0.f, 0.f, 0.f, 0.f};
    return z;
}

// A-operand (16x32 bf16, M x K): lane group g (=lane>>4) holds K-chunks
// [8g, 8g+8) and [16+8g, 16+8g+8) for row M = lane&15.  (ISA 7.12.2)
// Two b128 loads, joined via union (no element shuffling).
static __device__ __forceinline__ bf16x16 load_a16(const __bf16* __restrict__ p, int g) {
    ABReg u;
    u.h[0] = *(const bf16x8*)(p + 8 * g);
    u.h[1] = *(const bf16x8*)(p + 16 + 8 * g);
    return u.v;
}

// B-operand (32x16 bf16, K x N): lane group g holds K-chunks [8g, 8g+8) and
// [16+8g, 16+8g+8) for column N = lane&15 (same K order as A).  Two b128
// loads (16B-aligned: row strides 2048/4096B, k0*2 mult of 64B, 8*g*2 = 16B).
static __device__ __forceinline__ bf16x16 load_b16(const __bf16* __restrict__ p, int g) {
    ABReg u;
    u.h[0] = *(const bf16x8*)(p + 8 * g);
    u.h[1] = *(const bf16x8*)(p + 16 + 8 * g);
    return u.v;
}

static __device__ __forceinline__ f32x8 wmma_bf16(bf16x16 a, bf16x16 b, f32x8 c) {
    return __builtin_amdgcn_wmma_f32_16x16x32_bf16(false, a, false, b, (short)0, c,
                                                   false, false);
}

// ---------------------------------------------------------------- converts
__global__ void __launch_bounds__(256) cvt_f32_bf16(const float* __restrict__ in,
                                                    __bf16* __restrict__ out, int n) {
    int i = blockIdx.x * blockDim.x + threadIdx.x;
    int stride = gridDim.x * blockDim.x;
    for (; i < n; i += stride) out[i] = (__bf16)in[i];
}

// Wt[w][n][k] = W_w[k][n]  (transpose so B-operand reads are K-contiguous)
__global__ void __launch_bounds__(256) cvt_w_transpose(const float* __restrict__ Wq,
                                                       const float* __restrict__ Wk,
                                                       const float* __restrict__ Wv,
                                                       __bf16* __restrict__ Wt) {
    const int n = 3 * HH * CC;
    int i = blockIdx.x * blockDim.x + threadIdx.x;
    int stride = gridDim.x * blockDim.x;
    for (; i < n; i += stride) {
        int w = i / (HH * CC);
        int rem = i - w * (HH * CC);
        int col = rem / CC;   // output feature
        int k = rem - col * CC;
        const float* W = (w == 0) ? Wq : (w == 1) ? Wk : Wv;
        Wt[i] = (__bf16)W[(size_t)k * HH + col];
    }
}

// --------------------------------------------------- QKV projection (WMMA)
// One wave computes a 16(M) x 64(N) tile of Q, K and V at once (12 WMMAs per
// 32-wide k-chunk off a single A-tile load).
__global__ void __launch_bounds__(128, 1)
qkv_wmma(const __bf16* __restrict__ xb, const __bf16* __restrict__ Wt,
         const float* __restrict__ bq, const float* __restrict__ bk,
         const float* __restrict__ bv, __bf16* __restrict__ Qb,
         __bf16* __restrict__ Kb, __bf16* __restrict__ Vt) {
    int wave = blockIdx.x * 4 + (threadIdx.x >> 5);
    int lane = threadIdx.x & 31;
    int l16 = lane & 15, g = lane >> 4;

    const int nTiles = HH / 64;                 // 16
    int m0 = (wave / nTiles) * 16;              // 0..16383 (B*L rows)
    int n0 = (wave % nTiles) * 64;

    f32x8 acc[3][4];
#pragma unroll
    for (int w = 0; w < 3; ++w)
#pragma unroll
        for (int t = 0; t < 4; ++t) acc[w][t] = fzero8();

    const __bf16* arow = xb + (size_t)(m0 + l16) * CC;
    // Hoist the 12 weight-row pointers out of the k-loop.
    const __bf16* brow[3][4];
#pragma unroll
    for (int w = 0; w < 3; ++w)
#pragma unroll
        for (int t = 0; t < 4; ++t)
            brow[w][t] = Wt + ((size_t)w * HH + n0 + t * 16 + l16) * CC;

    for (int k0 = 0; k0 < CC; k0 += 32) {
        if (k0 + 32 < CC) __builtin_prefetch(arow + k0 + 32, 0, 0);
        bf16x16 A = load_a16(arow + k0, g);
#pragma unroll
        for (int w = 0; w < 3; ++w) {
#pragma unroll
            for (int t = 0; t < 4; ++t) {
                acc[w][t] = wmma_bf16(A, load_b16(brow[w][t] + k0, g), acc[w][t]);
            }
        }
    }

    int b = m0 >> 11;         // m0 / L
    int ml = m0 & (LL - 1);   // row within batch
#pragma unroll
    for (int t = 0; t < 4; ++t) {
        int n = n0 + t * 16 + l16;
        float bqv = bq[n], bkv = bk[n], bvv = bv[n];
#pragma unroll
        for (int e = 0; e < 8; ++e) {           // C layout: M = m0 + e + 8g
            int m = m0 + e + 8 * g;
            Qb[(size_t)m * HH + n] = (__bf16)(acc[0][t][e] + bqv);
            Kb[(size_t)m * HH + n] = (__bf16)(acc[1][t][e] + bkv);
        }
        bf16x8 vp;
#pragma unroll
        for (int e = 0; e < 8; ++e) vp[e] = (__bf16)(acc[2][t][e] + bvv);
        // V stored transposed [B, H, L]; 8 consecutive m per lane -> b128 store
        *(bf16x8*)(Vt + ((size_t)b * HH + n) * LL + ml + 8 * g) = vp;
    }
}

// --------------------------------------- scores + masked softmax -> P bf16
static __device__ __forceinline__ f32x8 score_tile(const __bf16* __restrict__ qrow,
                                                   const __bf16* __restrict__ krow, int g) {
    f32x8 a0 = fzero8(), a1 = fzero8();
#pragma unroll 2
    for (int h0 = 0; h0 < HH; h0 += 64) {
        a0 = wmma_bf16(load_a16(qrow + h0, g), load_b16(krow + h0, g), a0);
        a1 = wmma_bf16(load_a16(qrow + h0 + 32, g), load_b16(krow + h0 + 32, g), a1);
    }
#pragma unroll
    for (int e = 0; e < 8; ++e) a0[e] += a1[e];
    return a0;
}

__global__ void __launch_bounds__(128, 1)
attn_softmax(const __bf16* __restrict__ Qb, const __bf16* __restrict__ Kb,
             const int* __restrict__ mask, __bf16* __restrict__ P) {
    int wave = blockIdx.x * 4 + (threadIdx.x >> 5);
    int lane = threadIdx.x & 31;
    int l16 = lane & 15, g = lane >> 4;

    int b = wave / (LL / 16);
    int m0 = (wave % (LL / 16)) * 16;           // query tile within batch
    const float scale = 1.0f / 32.0f;           // 1/sqrt(1024)

    const __bf16* qrow = Qb + ((size_t)b * LL + m0 + l16) * HH;

    float mrun[8], lrun[8];
#pragma unroll
    for (int e = 0; e < 8; ++e) { mrun[e] = -3.0e38f; lrun[e] = 0.f; }

    // Pass 1: streaming online max / sum-exp over all key tiles
    for (int t = 0; t < LL / 16; ++t) {
        int n = t * 16 + l16;
        const __bf16* krow = Kb + ((size_t)b * LL + n) * HH;
        f32x8 s8 = score_tile(qrow, krow, g);
#pragma unroll
        for (int e = 0; e < 8; ++e) {
            int m = m0 + e + 8 * g;
            float s = s8[e] * scale;
            if (mask[(size_t)m * LL + n] != 0) s = -INFINITY;
            float mo = mrun[e];
            float mn = fmaxf(mo, s);
            lrun[e] = lrun[e] * __expf(mo - mn) + __expf(s - mn);
            mrun[e] = mn;
        }
    }
    // Cross-lane combine (rows live in 16-lane halves; xor 1..8 stays inside)
#pragma unroll
    for (int off = 1; off < 16; off <<= 1) {
#pragma unroll
        for (int e = 0; e < 8; ++e) {
            float mo = __shfl_xor(mrun[e], off, 32);
            float lo = __shfl_xor(lrun[e], off, 32);
            float mn = fmaxf(mrun[e], mo);
            lrun[e] = lrun[e] * __expf(mrun[e] - mn) + lo * __expf(mo - mn);
            mrun[e] = mn;
        }
    }
    float rinv[8];
#pragma unroll
    for (int e = 0; e < 8; ++e) rinv[e] = 1.0f / lrun[e];

    // Pass 2: recompute scores, write normalized probabilities as bf16
    for (int t = 0; t < LL / 16; ++t) {
        int n = t * 16 + l16;
        const __bf16* krow = Kb + ((size_t)b * LL + n) * HH;
        f32x8 s8 = score_tile(qrow, krow, g);
#pragma unroll
        for (int e = 0; e < 8; ++e) {
            int m = m0 + e + 8 * g;
            float s = s8[e] * scale;
            if (mask[(size_t)m * LL + n] != 0) s = -INFINITY;
            float p = __expf(s - mrun[e]) * rinv[e];
            P[((size_t)b * LL + m) * LL + n] = (__bf16)p;
        }
    }
}

// ------------------------------------------------------- O = P x V (WMMA)
__global__ void __launch_bounds__(128, 1)
out_wmma(const __bf16* __restrict__ P, const __bf16* __restrict__ Vt,
         float* __restrict__ out) {
    int wave = blockIdx.x * 4 + (threadIdx.x >> 5);
    int lane = threadIdx.x & 31;
    int l16 = lane & 15, g = lane >> 4;

    const int hTiles = HH / 64;                 // 16
    const int perB = (LL / 16) * hTiles;
    int b = wave / perB;
    int r = wave % perB;
    int m0 = (r / hTiles) * 16;
    int h0 = (r % hTiles) * 64;

    f32x8 acc[4];
#pragma unroll
    for (int t = 0; t < 4; ++t) acc[t] = fzero8();

    const __bf16* prow = P + ((size_t)b * LL + m0 + l16) * LL;
    const __bf16* vrow[4];
#pragma unroll
    for (int t = 0; t < 4; ++t)
        vrow[t] = Vt + ((size_t)b * HH + h0 + t * 16 + l16) * LL;

    for (int k0 = 0; k0 < LL; k0 += 32) {       // contraction over keys
        if (k0 + 32 < LL) __builtin_prefetch(prow + k0 + 32, 0, 0);
        bf16x16 A = load_a16(prow + k0, g);
#pragma unroll
        for (int t = 0; t < 4; ++t) {
            acc[t] = wmma_bf16(A, load_b16(vrow[t] + k0, g), acc[t]);
        }
    }
#pragma unroll
    for (int t = 0; t < 4; ++t) {
        int h = h0 + t * 16 + l16;
#pragma unroll
        for (int e = 0; e < 8; ++e) {
            int m = m0 + e + 8 * g;
            out[((size_t)b * LL + m) * HH + h] = acc[t][e];
        }
    }
}

// ------------------------------------------------------------------ launch
extern "C" void kernel_launch(void* const* d_in, const int* in_sizes, int n_in,
                              void* d_out, int out_size, void* d_ws, size_t ws_size,
                              hipStream_t stream) {
    const float* x  = (const float*)d_in[0];
    const float* Wq = (const float*)d_in[1];
    const float* bq = (const float*)d_in[2];
    const float* Wk = (const float*)d_in[3];
    const float* bk = (const float*)d_in[4];
    const float* Wv = (const float*)d_in[5];
    const float* bv = (const float*)d_in[6];
    const int*   mask = (const int*)d_in[7];
    float* out = (float*)d_out;

    // Workspace layout (bf16 elements), ~208 MB total.
    __bf16* ws = (__bf16*)d_ws;
    __bf16* xb = ws;                                    // B*L*C
    __bf16* Wt = xb + (size_t)BB * LL * CC;             // 3*H*C
    __bf16* Qb = Wt + (size_t)3 * HH * CC;              // B*L*H
    __bf16* Kb = Qb + (size_t)BB * LL * HH;             // B*L*H
    __bf16* Vt = Kb + (size_t)BB * LL * HH;             // B*H*L (transposed)
    __bf16* P  = Vt + (size_t)BB * HH * LL;             // B*L*L

    cvt_f32_bf16<<<4096, 256, 0, stream>>>(x, xb, BB * LL * CC);
    cvt_w_transpose<<<3072, 256, 0, stream>>>(Wq, Wk, Wv, Wt);

    // QKV: (B*L/16) * (H/64) waves, 4 waves / block
    qkv_wmma<<<((BB * LL / 16) * (HH / 64)) / 4, 128, 0, stream>>>(xb, Wt, bq, bk, bv,
                                                                   Qb, Kb, Vt);
    // Softmax: B * (L/16) waves
    attn_softmax<<<(BB * (LL / 16)) / 4, 128, 0, stream>>>(Qb, Kb, mask, P);

    // Output: (B*L/16) * (H/64) waves
    out_wmma<<<((BB * LL / 16) * (HH / 64)) / 4, 128, 0, stream>>>(P, Vt, out);
}
